// GeneralizedMatrix_76622216561027
// MI455X (gfx1250) — matrix operand order repaired and checked by reading, not compiled, not run on hardware
//
#include <hip/hip_runtime.h>
#include <hip/hip_bf16.h>

// ---- problem constants (match reference) ----
#define IN_F    4096
#define OUT_F   4096
#define BSZ     128
#define NI      32
#define NO      32
#define ABI     8
#define LOWRANK 64

// LDS strides (bf16 elements); 136*2 = 272 bytes per row -> 16B aligned rows
#define AST 136
#define BST 136

typedef __attribute__((ext_vector_type(8)))  float   v8f;
typedef __attribute__((ext_vector_type(16))) __bf16  v16bf;
typedef __attribute__((ext_vector_type(8)))  __bf16  bf16x8;

union frag16 { bf16x8 h[2]; v16bf v; };

__device__ __forceinline__ void cvt4(__bf16* dst, float4 v) {
    dst[0] = (__bf16)v.x; dst[1] = (__bf16)v.y;
    dst[2] = (__bf16)v.z; dst[3] = (__bf16)v.w;
}

// ---------------------------------------------------------------------------
// Kernel A: h = x @ lowrank_first^T   (M=nrows, N=64, K=4096), h stored bf16.
// Block: 128 threads (4 waves). Tile: 64 rows x 64 cols. K staged 128/iter.
// ---------------------------------------------------------------------------
__global__ __launch_bounds__(128) void lowrank_h_kernel(
    const float* __restrict__ x, const float* __restrict__ l1,
    __bf16* __restrict__ hws, int nrows)
{
    __shared__ __attribute__((aligned(16))) __bf16 Alds[64 * AST];
    __shared__ __attribute__((aligned(16))) __bf16 Blds[64 * BST];

    const int mb   = blockIdx.x * 64;
    const int tid  = threadIdx.x;
    const int lane = tid & 31;
    const int wave = tid >> 5;         // 0..3
    const int m0   = wave * 16;        // 16-row strip per wave
    const int lrow = lane & 15;
    const int hi   = lane >> 4;

    v8f acc[4] = {};                   // 4 N-tiles of 16

    for (int kc = 0; kc < IN_F; kc += 128) {
        // Stage A: x[mb..mb+63][kc..kc+127] -> bf16
        #pragma unroll
        for (int p = 0; p < 16; ++p) {
            int idx = p * 128 + tid;           // 2048 float4 chunks
            int r   = idx >> 5;                // 0..63
            int k4  = (idx & 31) * 4;          // 0..124
            float4 v = *(const float4*)(x + (size_t)(mb + r) * IN_F + kc + k4);
            cvt4(&Alds[r * AST + k4], v);
        }
        // Stage B (already "transposed"): Blds[n][k] = l1[n*IN_F + kc + k]
        #pragma unroll
        for (int p = 0; p < 16; ++p) {
            int idx = p * 128 + tid;
            int r   = idx >> 5;                // n: 0..63
            int k4  = (idx & 31) * 4;
            float4 v = *(const float4*)(l1 + (size_t)r * IN_F + kc + k4);
            cvt4(&Blds[r * BST + k4], v);
        }
        __syncthreads();

        #pragma unroll
        for (int ks = 0; ks < 4; ++ks) {
            const int k0 = ks * 32;
            frag16 af;
            const __bf16* ap = &Alds[(m0 + lrow) * AST + k0 + hi * 8];
            af.h[0] = *(const bf16x8*)ap;
            af.h[1] = *(const bf16x8*)(ap + 16);
            #pragma unroll
            for (int nt = 0; nt < 4; ++nt) {
                frag16 bfr;
                const __bf16* bp = &Blds[(nt * 16 + lrow) * BST + k0 + hi * 8];
                bfr.h[0] = *(const bf16x8*)bp;
                bfr.h[1] = *(const bf16x8*)(bp + 16);
                acc[nt] = __builtin_amdgcn_wmma_f32_16x16x32_bf16(
                    false, af.v, false, bfr.v, (short)0, acc[nt], false, false);
            }
        }
        __syncthreads();
    }

    // Store h as bf16: h[row][r], row = mb + m0 + hi*8 + v, r = nt*16 + lrow
    #pragma unroll
    for (int nt = 0; nt < 4; ++nt) {
        #pragma unroll
        for (int v = 0; v < 8; ++v) {
            int row = mb + m0 + hi * 8 + v;
            hws[(size_t)row * LOWRANK + nt * 16 + lrow] = (__bf16)acc[nt][v];
        }
    }
}

// ---------------------------------------------------------------------------
// Kernel B: block-sparse matmul + low-rank add.
// Grid: (NO, nrows/128). Block: 256 threads (8 waves).
// Tile: 128 rows x 128 cols (one output block). Each wave: 32 rows x 64 cols.
// ---------------------------------------------------------------------------
__global__ __launch_bounds__(256) void bsmm_kernel(
    const float* __restrict__ x, const float* __restrict__ w,
    const float* __restrict__ l2, const __bf16* __restrict__ hws,
    const int* __restrict__ fwd, float* __restrict__ out, int nrows)
{
    __shared__ __attribute__((aligned(16))) __bf16 Alds[128 * AST];
    __shared__ __attribute__((aligned(16))) __bf16 Blds[128 * BST];
    __shared__ int ilist[ABI];
    __shared__ int alist[ABI];
    __shared__ int cnt_s;

    const int j    = blockIdx.x;           // output block
    const int mb   = blockIdx.y * 128;     // row tile base
    const int tid  = threadIdx.x;
    const int lane = tid & 31;
    const int wave = tid >> 5;             // 0..7
    const int wm   = (wave & 3) * 32;      // wave row offset within tile
    const int wn   = (wave >> 2) * 64;     // wave col offset within tile
    const int lrow = lane & 15;
    const int hi   = lane >> 4;

    // Deterministic inverse-index scan: which (i, a) feed output block j.
    if (tid == 0) {
        int c = 0;
        for (int t = 0; t < NI * ABI; ++t)
            if (fwd[t] == j && c < ABI) { ilist[c] = t >> 3; alist[c] = t & 7; ++c; }
        cnt_s = c;
    }
    __syncthreads();
    const int cnt = cnt_s;

    v8f acc[2][4] = {};

    for (int b = 0; b < cnt; ++b) {
        const int i = ilist[b], a = alist[b];

        // Stage x tile: Alds[r][k] = x[mb+r][i*128 + k]   (fp32 -> bf16)
        const float* xsrc = x + (size_t)mb * IN_F + (size_t)i * BSZ;
        #pragma unroll
        for (int p = 0; p < 16; ++p) {
            int r  = p * 8 + (tid >> 5);       // 0..127
            int k4 = (tid & 31) * 4;           // 0..124
            float4 v = *(const float4*)(xsrc + (size_t)r * IN_F + k4);
            cvt4(&Alds[r * AST + k4], v);
        }
        // Stage W tile transposed: Blds[c][k] = w[(i*128+k)*1024 + a*128 + c]
        const float* wsrc = w + (size_t)i * BSZ * (ABI * BSZ) + (size_t)a * BSZ;
        #pragma unroll
        for (int p = 0; p < 16; ++p) {
            int k  = p * 8 + (tid >> 5);
            int c4 = (tid & 31) * 4;
            float4 v = *(const float4*)(wsrc + (size_t)k * (ABI * BSZ) + c4);
            Blds[(c4 + 0) * BST + k] = (__bf16)v.x;
            Blds[(c4 + 1) * BST + k] = (__bf16)v.y;
            Blds[(c4 + 2) * BST + k] = (__bf16)v.z;
            Blds[(c4 + 3) * BST + k] = (__bf16)v.w;
        }
        __syncthreads();

        // Warm L2 with next weight tile while we compute this one.
        if (b + 1 < cnt) {
            const float* nw = w + (size_t)ilist[b + 1] * BSZ * (ABI * BSZ)
                                + (size_t)alist[b + 1] * BSZ;
            __builtin_prefetch(nw + (size_t)(tid >> 1) * (ABI * BSZ) + (tid & 1) * 64, 0, 1);
        }

        #pragma unroll
        for (int ks = 0; ks < 4; ++ks) {
            const int k0 = ks * 32;
            frag16 af[2];
            #pragma unroll
            for (int m16 = 0; m16 < 2; ++m16) {
                const __bf16* ap = &Alds[(wm + m16 * 16 + lrow) * AST + k0 + hi * 8];
                af[m16].h[0] = *(const bf16x8*)ap;
                af[m16].h[1] = *(const bf16x8*)(ap + 16);
            }
            #pragma unroll
            for (int nt = 0; nt < 4; ++nt) {
                frag16 bfr;
                const __bf16* bp = &Blds[(wn + nt * 16 + lrow) * BST + k0 + hi * 8];
                bfr.h[0] = *(const bf16x8*)bp;
                bfr.h[1] = *(const bf16x8*)(bp + 16);
                #pragma unroll
                for (int m16 = 0; m16 < 2; ++m16) {
                    acc[m16][nt] = __builtin_amdgcn_wmma_f32_16x16x32_bf16(
                        false, af[m16].v, false, bfr.v, (short)0,
                        acc[m16][nt], false, false);
                }
            }
        }
        __syncthreads();
    }

    // ---- Low-rank correction: out_tile += h_tile[128x64] @ L2_block^T ----
    {
        // Stage h tile (already bf16): Alds[r][0..63]
        const __bf16* hsrc = hws + (size_t)mb * LOWRANK;
        #pragma unroll
        for (int p = 0; p < 4; ++p) {
            int idx = p * 256 + tid;            // 1024 chunks of 8 bf16
            int r   = idx >> 3;                 // 0..127
            int c8  = (idx & 7) * 8;            // 0..56
            *(bf16x8*)&Alds[r * AST + c8] =
                *(const bf16x8*)(hsrc + (size_t)r * LOWRANK + c8);
        }
        // Stage L2 block transposed: Blds[c][r] = l2[(j*128+c)*64 + r]
        const float* lsrc = l2 + (size_t)j * BSZ * LOWRANK;
        #pragma unroll
        for (int p = 0; p < 8; ++p) {
            int idx = p * 256 + tid;            // 2048 float4 chunks
            int c   = idx >> 4;                 // 0..127
            int r4  = (idx & 15) * 4;           // 0..60
            float4 v = *(const float4*)(lsrc + (size_t)c * LOWRANK + r4);
            cvt4(&Blds[c * BST + r4], v);
        }
    }
    __syncthreads();

    #pragma unroll
    for (int ks = 0; ks < 2; ++ks) {
        const int k0 = ks * 32;
        frag16 af[2];
        #pragma unroll
        for (int m16 = 0; m16 < 2; ++m16) {
            const __bf16* ap = &Alds[(wm + m16 * 16 + lrow) * AST + k0 + hi * 8];
            af[m16].h[0] = *(const bf16x8*)ap;
            af[m16].h[1] = *(const bf16x8*)(ap + 16);
        }
        #pragma unroll
        for (int nt = 0; nt < 4; ++nt) {
            frag16 bfr;
            const __bf16* bp = &Blds[(wn + nt * 16 + lrow) * BST + k0 + hi * 8];
            bfr.h[0] = *(const bf16x8*)bp;
            bfr.h[1] = *(const bf16x8*)(bp + 16);
            #pragma unroll
            for (int m16 = 0; m16 < 2; ++m16) {
                acc[m16][nt] = __builtin_amdgcn_wmma_f32_16x16x32_bf16(
                    false, af[m16].v, false, bfr.v, (short)0,
                    acc[m16][nt], false, false);
            }
        }
    }

    // ---- Epilogue: write f32 result (covers every output element) ----
    float* obase = out + (size_t)mb * OUT_F + (size_t)j * BSZ;
    #pragma unroll
    for (int m16 = 0; m16 < 2; ++m16) {
        #pragma unroll
        for (int nt = 0; nt < 4; ++nt) {
            int col = wn + nt * 16 + lrow;
            #pragma unroll
            for (int v = 0; v < 8; ++v) {
                int row = wm + m16 * 16 + hi * 8 + v;
                obase[(size_t)row * OUT_F + col] = acc[m16][nt][v];
            }
        }
    }
}

// ---------------------------------------------------------------------------
extern "C" void kernel_launch(void* const* d_in, const int* in_sizes, int n_in,
                              void* d_out, int out_size, void* d_ws, size_t ws_size,
                              hipStream_t stream)
{
    const float* x   = (const float*)d_in[0];   // [nrows, 4096]
    const float* w   = (const float*)d_in[1];   // [4096, 8, 128]
    const float* l1  = (const float*)d_in[2];   // [64, 4096]
    const float* l2  = (const float*)d_in[3];   // [4096, 64]
    const int*   fwd = (const int*)d_in[4];     // [32, 8]
    float*       out = (float*)d_out;

    const int nrows = in_sizes[0] / IN_F;       // 8192
    __bf16* hws = (__bf16*)d_ws;                // nrows*64 bf16 = 1 MB

    dim3 gA(nrows / 64);
    lowrank_h_kernel<<<gA, 128, 0, stream>>>(x, l1, hws, nrows);

    dim3 gB(NO, nrows / 128);
    bsmm_kernel<<<gB, 256, 0, stream>>>(x, w, l2, hws, fwd, out, nrows);
}
